// SimpleMinGRU_20383914787425
// MI455X (gfx1250) — matrix operand order repaired and checked by reading, not compiled, not run on hardware
//
#include <hip/hip_runtime.h>
#include <hip/hip_bf16.h>

// ---------------------------------------------------------------------------
// SimpleMinGRU on MI455X (gfx1250).
//   B=64, L=2048, H=512.  Key insight: inp = x*W_in + b_in is rank-1, so the
//   z / h_tilde GEMMs collapse to per-channel affine maps of the scalar x.
//   Pipeline:
//     1) prep:   u_z = W_z@W_in, v_z = W_z@b_in + b_z  (and same for W_h)
//     2) cvt:    W_c -> bf16
//     3) scan:   h_t = (1-z_t) h_{t-1} + z_t tanh(...), h stored as bf16
//     4) gemm:   flow_ctx = h @ W_c^T + b_c  via v_wmma_f32_16x16x32_bf16
//     5) recon:  out = h @ W_g^T + b_g       (N=2, wave-per-row dot products)
// ---------------------------------------------------------------------------

typedef __bf16 bf16;
typedef bf16  v16bf __attribute__((ext_vector_type(16)));
typedef bf16  v8bf  __attribute__((ext_vector_type(8)));
typedef float v8f   __attribute__((ext_vector_type(8)));

__device__ __forceinline__ unsigned short f2bf(float f) {
    unsigned u = __float_as_uint(f);
    unsigned r = u + 0x7FFFu + ((u >> 16) & 1u);   // round-to-nearest-even
    return (unsigned short)(r >> 16);
}
__device__ __forceinline__ float bf2f(unsigned short s) {
    return __uint_as_float(((unsigned)s) << 16);
}

#define MG_B 64
#define MG_L 2048
#define MG_H 512
#define MG_M (MG_B * MG_L)   // 131072 rows

// ---------------------------------------------------------------------------
// 1) fold rank-1 input projection into the gate weights
//    u[k] = sum_h W[k,h]*W_in[h] ; v[k] = sum_h W[k,h]*b_in[h] + bias[k]
// ---------------------------------------------------------------------------
__global__ __launch_bounds__(256) void mg_prep(
    const float* __restrict__ Win, const float* __restrict__ bin,
    const float* __restrict__ Wz,  const float* __restrict__ bz,
    const float* __restrict__ Wh,  const float* __restrict__ bh,
    float* __restrict__ uz, float* __restrict__ vz,
    float* __restrict__ uh, float* __restrict__ vh)
{
    int k = blockIdx.x * 256 + threadIdx.x;
    if (k >= MG_H) return;
    const float* rz = Wz + (size_t)k * MG_H;
    const float* rh = Wh + (size_t)k * MG_H;
    float suz = 0.f, svz = 0.f, suh = 0.f, svh = 0.f;
    for (int h = 0; h < MG_H; ++h) {
        float wi = Win[h], bi = bin[h];
        suz = fmaf(rz[h], wi, suz);  svz = fmaf(rz[h], bi, svz);
        suh = fmaf(rh[h], wi, suh);  svh = fmaf(rh[h], bi, svh);
    }
    uz[k] = suz;  vz[k] = svz + bz[k];
    uh[k] = suh;  vh[k] = svh + bh[k];
}

// ---------------------------------------------------------------------------
// 2) f32 -> bf16 weight conversion
// ---------------------------------------------------------------------------
__global__ __launch_bounds__(256) void mg_cvt(
    const float* __restrict__ src, unsigned short* __restrict__ dst, int n)
{
    int i = blockIdx.x * 256 + threadIdx.x;
    if (i < n) dst[i] = f2bf(src[i]);
}

// ---------------------------------------------------------------------------
// 3) linear recurrence: one thread owns channel (b,k) across all t.
//    z,h_tilde depend only on x[b,t] (rank-1 collapse). h kept in f32,
//    written out as bf16 (GEMM operand precision).
// ---------------------------------------------------------------------------
__global__ __launch_bounds__(256) void mg_scan(
    const float* __restrict__ x,
    const float* __restrict__ uz, const float* __restrict__ vz,
    const float* __restrict__ uh, const float* __restrict__ vh,
    unsigned short* __restrict__ hout)
{
    int k = (blockIdx.x & 1) * 256 + threadIdx.x;   // 0..H-1
    int b = blockIdx.x >> 1;
    const float uzk = uz[k], vzk = vz[k], uhk = uh[k], vhk = vh[k];
    const float* xr = x + (size_t)b * MG_L;
    unsigned short* orow = hout + (size_t)b * MG_L * MG_H + k;
    float h = 0.f;
    for (int t = 0; t < MG_L; ++t) {
        float xv = xr[t];                       // broadcast across the block
        float z  = 1.f / (1.f + __expf(-(fmaf(xv, uzk, vzk))));
        float ht = 2.f / (1.f + __expf(-2.f * fmaf(xv, uhk, vhk))) - 1.f;
        h = fmaf(z, ht - h, h);                 // (1-z)*h + z*ht
        orow[(size_t)t * MG_H] = f2bf(h);
    }
}

// ---------------------------------------------------------------------------
// 4) flow_ctx = h @ W_c^T + b_c  --  bf16 WMMA, f32 accumulate.
//    Block: 256 thr (8 waves), tile 64(M) x 128(N); wave tile 32x32 (2x2 WMMA).
//    B-matrix column n == W_c row n (contiguous over K): one aligned v16bf
//    load per B fragment. Bias folded into accumulator init (per-column).
// ---------------------------------------------------------------------------
__global__ __launch_bounds__(256) void mg_gemm_flow(
    const unsigned short* __restrict__ hbf,    // M x K bf16
    const unsigned short* __restrict__ wcbf,   // N x K bf16
    const float* __restrict__ bias,            // N
    float* __restrict__ out)                   // M x N f32
{
    const int K = MG_H, N = MG_H;
    const int tid  = threadIdx.x;
    const int lane = tid & 31;
    const int w    = tid >> 5;                 // 0..7
    const int wm   = w & 1;                    // 2 waves along M
    const int wn   = w >> 1;                   // 4 waves along N
    const int mW   = blockIdx.x * 64  + wm * 32;
    const int nW   = blockIdx.y * 128 + wn * 32;
    const int l15  = lane & 15;
    const int lhi  = lane >> 4;                // 0 | 1

    const bf16* A  = reinterpret_cast<const bf16*>(hbf);
    const bf16* Bm = reinterpret_cast<const bf16*>(wcbf);

    // per-lane base rows
    const size_t ar0 = (size_t)(mW + l15)      * K;
    const size_t ar1 = (size_t)(mW + 16 + l15) * K;
    const size_t br0 = (size_t)(nW + l15)      * K;
    const size_t br1 = (size_t)(nW + 16 + l15) * K;
    const int kA = lhi * 16;                   // A: K chunks [kA..kA+7],[kA+8..kA+15]
    const int kB = lhi * 16;                   // B: K chunk  [kB..kB+15]

    const float bv0 = bias[nW + l15];
    const float bv1 = bias[nW + 16 + l15];
    v8f acc00, acc01, acc10, acc11;
#pragma unroll
    for (int i = 0; i < 8; ++i) {
        acc00[i] = bv0; acc10[i] = bv0;        // tj = 0 columns
        acc01[i] = bv1; acc11[i] = bv1;        // tj = 1 columns
    }

#pragma unroll 4
    for (int k = 0; k < K; k += 32) {
        // A fragments (two 16B loads each, contiguous k = 16*half + s, same order as B)
        const bf16* pa0 = A + ar0 + k + kA;
        const bf16* pa1 = A + ar1 + k + kA;
        v8bf a0lo = *reinterpret_cast<const v8bf*>(pa0);
        v8bf a0hi = *reinterpret_cast<const v8bf*>(pa0 + 8);
        v8bf a1lo = *reinterpret_cast<const v8bf*>(pa1);
        v8bf a1hi = *reinterpret_cast<const v8bf*>(pa1 + 8);
        v16bf a0 = __builtin_shufflevector(a0lo, a0hi,
                    0,1,2,3,4,5,6,7,8,9,10,11,12,13,14,15);
        v16bf a1 = __builtin_shufflevector(a1lo, a1hi,
                    0,1,2,3,4,5,6,7,8,9,10,11,12,13,14,15);
        // B fragments (single 32B-aligned load: 16 consecutive K of W_c row)
        v16bf b0 = *reinterpret_cast<const v16bf*>(Bm + br0 + k + kB);
        v16bf b1 = *reinterpret_cast<const v16bf*>(Bm + br1 + k + kB);

        acc00 = __builtin_amdgcn_wmma_f32_16x16x32_bf16(
                    false, a0, false, b0, (short)0, acc00, false, false);
        acc01 = __builtin_amdgcn_wmma_f32_16x16x32_bf16(
                    false, a0, false, b1, (short)0, acc01, false, false);
        acc10 = __builtin_amdgcn_wmma_f32_16x16x32_bf16(
                    false, a1, false, b0, (short)0, acc10, false, false);
        acc11 = __builtin_amdgcn_wmma_f32_16x16x32_bf16(
                    false, a1, false, b1, (short)0, acc11, false, false);
    }

    // D layout: VGPR v, lanes 0-15 -> M=v, lanes 16-31 -> M=v+8; col = lane&15
    const int rb = 8 * lhi;
#pragma unroll
    for (int v = 0; v < 8; ++v) {
        size_t r0 = (size_t)(mW +      rb + v) * N;
        size_t r1 = (size_t)(mW + 16 + rb + v) * N;
        out[r0 + nW + l15]      = acc00[v];
        out[r0 + nW + 16 + l15] = acc01[v];
        out[r1 + nW + l15]      = acc10[v];
        out[r1 + nW + 16 + l15] = acc11[v];
    }
}

// ---------------------------------------------------------------------------
// 5) recon = h @ W_g^T + b_g  (N=2). One wave per output row; lanes stride K,
//    coalesced 64B bf16 reads, wave32 shuffle reduction.
// ---------------------------------------------------------------------------
__global__ __launch_bounds__(256) void mg_recon(
    const unsigned short* __restrict__ hbf,
    const float* __restrict__ Wg, const float* __restrict__ bg,
    float* __restrict__ out)
{
    int row  = (int)((blockIdx.x * 256 + threadIdx.x) >> 5);
    int lane = threadIdx.x & 31;
    if (row >= MG_M) return;
    const unsigned short* hr = hbf + (size_t)row * MG_H;
    float s0 = 0.f, s1 = 0.f;
    for (int k = lane; k < MG_H; k += 32) {
        float hv = bf2f(hr[k]);
        s0 = fmaf(hv, Wg[k],        s0);
        s1 = fmaf(hv, Wg[MG_H + k], s1);
    }
#pragma unroll
    for (int off = 16; off > 0; off >>= 1) {
        s0 += __shfl_down(s0, off, 32);
        s1 += __shfl_down(s1, off, 32);
    }
    if (lane == 0) {
        out[(size_t)row * 2 + 0] = s0 + bg[0];
        out[(size_t)row * 2 + 1] = s1 + bg[1];
    }
}

// ---------------------------------------------------------------------------
extern "C" void kernel_launch(void* const* d_in, const int* in_sizes, int n_in,
                              void* d_out, int out_size, void* d_ws, size_t ws_size,
                              hipStream_t stream) {
    (void)in_sizes; (void)n_in; (void)out_size; (void)ws_size;
    // setup_inputs order: x, W_in, b_in, W_z, b_z, W_h, b_h, W_g, b_g, W_c, b_c
    const float* x    = (const float*)d_in[0];
    const float* W_in = (const float*)d_in[1];
    const float* b_in = (const float*)d_in[2];
    const float* W_z  = (const float*)d_in[3];
    const float* b_z  = (const float*)d_in[4];
    const float* W_h  = (const float*)d_in[5];
    const float* b_h  = (const float*)d_in[6];
    const float* W_g  = (const float*)d_in[7];
    const float* b_g  = (const float*)d_in[8];
    const float* W_c  = (const float*)d_in[9];
    const float* b_c  = (const float*)d_in[10];

    // workspace layout
    char* ws = (char*)d_ws;
    float* uz = (float*)(ws + 0);
    float* vz = (float*)(ws + 2048);
    float* uh = (float*)(ws + 4096);
    float* vh = (float*)(ws + 6144);
    unsigned short* wc_bf = (unsigned short*)(ws + 8192);                // 512 KB
    unsigned short* h_bf  = (unsigned short*)(ws + 8192 + 524288);      // 128 MB

    float* recon = (float*)d_out;                       // (B,L,2)
    float* flow  = (float*)d_out + (size_t)MG_M * 2;    // (B,L,H)

    mg_prep<<<2, 256, 0, stream>>>(W_in, b_in, W_z, b_z, W_h, b_h,
                                   uz, vz, uh, vh);
    mg_cvt<<<(MG_H * MG_H + 255) / 256, 256, 0, stream>>>(W_c, wc_bf,
                                                          MG_H * MG_H);
    mg_scan<<<MG_B * 2, 256, 0, stream>>>(x, uz, vz, uh, vh, h_bf);
    mg_gemm_flow<<<dim3(MG_M / 64, MG_H / 128), 256, 0, stream>>>(
        h_bf, wc_bf, b_c, flow);
    mg_recon<<<MG_M / 8, 256, 0, stream>>>(h_bf, W_g, b_g, recon);
}
